// GeoHRSNN_68753836474610
// MI455X (gfx1250) — matrix operand order repaired and checked by reading, not compiled, not run on hardware
//
#include <hip/hip_runtime.h>
#include <hip/hip_bf16.h>

// ---------------------------------------------------------------------------
// Problem constants (match reference setup_inputs)
// ---------------------------------------------------------------------------
#define BATCH 4096
#define TSTEPS 320
#define HDIM 64          // IN == H == 64
#define DPW 10           // T / 32 windows
#define DPF 640          // DPW * H
#define DCAT 672         // 640 + 32

typedef __attribute__((ext_vector_type(16))) __bf16 v16bf;
typedef __attribute__((ext_vector_type(8)))  __bf16 v8bf;
typedef __attribute__((ext_vector_type(8)))  float  v8f;
typedef __attribute__((ext_vector_type(4)))  float  v4f;

__device__ __forceinline__ __bf16 tobf(float f) { return (__bf16)f; }

__device__ __forceinline__ v8f wmma_bf16(v16bf a, v16bf b, v8f c) {
    // (neg_a, A, neg_b, B, c_mod, C, reuse_a, reuse_b)
    return __builtin_amdgcn_wmma_f32_16x16x32_bf16(false, a, false, b,
                                                   (short)0, c, false, false);
}

// ---------------------------------------------------------------------------
// Kernel 1: fused spiking-scan.
// One wave owns a 16-row batch tile. Per timestep:
//   cur    = bf16-WMMA( x_tile , Wsf )        (2 K-tiles x 4 N-tiles)
//   spikes = elementwise on C-layout registers
//   cur_li = bf16-WMMA( [sf|ss], Wli )        (4 K-tiles x 4 N-tiles)
//   ml     = 0.9*ml + cur_li ;  windowed +- accumulation into acc
// Spikes are staged C-layout -> LDS -> A-layout (per-wave buffer).
// ---------------------------------------------------------------------------
__global__ __launch_bounds__(128) void snn_scan_kernel(
    const float* __restrict__ x,    // [B,T,64]
    const float* __restrict__ Wsf,  // [64,64]
    const float* __restrict__ bsf,  // [64]
    const float* __restrict__ Wli,  // [128,64]
    const float* __restrict__ bli,  // [64]
    float* __restrict__ dp)         // [B,640]
{
    // B-matrix fragments (bf16), one v16bf per lane per 32x16 tile.
    __shared__ v16bf wsfF[8][32];    // 2 Kb x 4 Nt
    __shared__ v16bf wliF[16][32];   // 4 Kb x 4 Nt
    __shared__ __bf16 spk[4][16][128];  // per-wave spike staging, rows x K

    const int tid  = threadIdx.x;
    const int wv   = tid >> 5;
    const int l    = tid & 31;
    const int nloc = l & 15;     // column within 16-tile / row for A reads
    const int hh   = l >> 4;     // lane half

    // ---- pack Wsf into bf16 B-fragments: element j <-> K = kb*32 + (j<8 ? 8*hh + j : 16 + 8*hh + j-8)
    for (int tile = wv; tile < 8; tile += 4) {
        const int kb = tile >> 2, nt = tile & 3;
        const int n = nt * 16 + nloc;
        v16bf f;
#pragma unroll
        for (int j = 0; j < 16; ++j) {
            const int k = kb * 32 + (j < 8 ? 8 * hh + j : 16 + 8 * hh + (j - 8));
            f[j] = tobf(Wsf[k * HDIM + n]);
        }
        wsfF[tile][l] = f;
    }
    // ---- pack Wli ([128,64]) the same way
    for (int tile = wv; tile < 16; tile += 4) {
        const int kb = tile >> 2, nt = tile & 3;
        const int n = nt * 16 + nloc;
        v16bf f;
#pragma unroll
        for (int j = 0; j < 16; ++j) {
            const int k = kb * 32 + (j < 8 ? 8 * hh + j : 16 + 8 * hh + (j - 8));
            f[j] = tobf(Wli[k * HDIM + n]);
        }
        wliF[tile][l] = f;
    }
    __syncthreads();

    // per-lane bias values (one column per lane per N-tile)
    float bsfv[4], bliv[4];
#pragma unroll
    for (int nt = 0; nt < 4; ++nt) {
        bsfv[nt] = bsf[nt * 16 + nloc];
        bliv[nt] = bli[nt * 16 + nloc];
    }

    // membrane / accumulator state, C-matrix layout: elem i <-> row i+8*hh
    v8f mf[4], ms[4], ml[4], acc[4];
#pragma unroll
    for (int nt = 0; nt < 4; ++nt)
#pragma unroll
        for (int i = 0; i < 8; ++i) {
            mf[nt][i] = 0.f; ms[nt][i] = 0.f; ml[nt][i] = 0.f; acc[nt][i] = 0.f;
        }

    const int b0 = (blockIdx.x * 4 + wv) * 16;
    const float* xrow = x + ((size_t)(b0 + nloc) * TSTEPS) * HDIM;

    for (int t = 0; t < TSTEPS; ++t) {
        // ---- load x A-fragments straight from global in A layout:
        //  lane (row=nloc, half=hh): chunk0 = K in [ka*32+8*hh, +8)
        //                            chunk1 = chunk0 + 16
        v16bf ax[2];
#pragma unroll
        for (int ka = 0; ka < 2; ++ka) {
            const float* base = xrow + (size_t)t * HDIM + ka * 32 + 8 * hh;
            if (t + 1 < TSTEPS) __builtin_prefetch(base + HDIM, 0, 1);
            v4f c0a = *(const v4f*)(base);
            v4f c0b = *(const v4f*)(base + 4);
            v4f c1a = *(const v4f*)(base + 16);
            v4f c1b = *(const v4f*)(base + 20);
            v16bf a;
#pragma unroll
            for (int j = 0; j < 4; ++j) {
                a[j]      = tobf(c0a[j]);
                a[4 + j]  = tobf(c0b[j]);
                a[8 + j]  = tobf(c1a[j]);
                a[12 + j] = tobf(c1b[j]);
            }
            ax[ka] = a;
        }

        // ---- cur = x@Wsf + bsf ; membrane update + spike emit (to LDS)
#pragma unroll
        for (int nt = 0; nt < 4; ++nt) {
            v8f c;
#pragma unroll
            for (int i = 0; i < 8; ++i) c[i] = bsfv[nt];
            c = wmma_bf16(ax[0], wsfF[0 * 4 + nt][l], c);
            c = wmma_bf16(ax[1], wsfF[1 * 4 + nt][l], c);
#pragma unroll
            for (int i = 0; i < 8; ++i) {
                const float cur = c[i];
                float vmf = 0.5f * mf[nt][i] + cur;
                const float sf = (vmf >= 0.5f) ? 1.0f : 0.0f;
                mf[nt][i] = vmf * (1.0f - sf);
                float vms = 0.9f * ms[nt][i] + cur;
                const float ss = (vms >= 1.0f) ? 1.0f : 0.0f;
                ms[nt][i] = vms * (1.0f - ss);
                const int m = i + 8 * hh;
                const int n = nt * 16 + nloc;
                spk[wv][m][n]      = tobf(sf);
                spk[wv][m][64 + n] = tobf(ss);
            }
        }

        // per-wave staging buffer; DS ops of a wave are serviced in order,
        // but make the store->load ordering explicit:
        asm volatile("s_wait_dscnt 0" ::: "memory");

        // ---- read spike A-fragments (row = nloc, K = kt*32 + {8*hh, 16+8*hh})
        v16bf as[4];
#pragma unroll
        for (int kt = 0; kt < 4; ++kt) {
            const __bf16* row = &spk[wv][nloc][0];
            v8bf lo = *(const v8bf*)(row + kt * 32 + 8 * hh);
            v8bf hi = *(const v8bf*)(row + kt * 32 + 16 + 8 * hh);
            as[kt] = __builtin_shufflevector(lo, hi,
                     0, 1, 2, 3, 4, 5, 6, 7, 8, 9, 10, 11, 12, 13, 14, 15);
        }

        // ---- cur_li = [sf|ss]@Wli + bli ; leaky integrate ; window +-
        const float sgn = ((t & 31) >= 16) ? 1.0f : -1.0f;
#pragma unroll
        for (int nt = 0; nt < 4; ++nt) {
            v8f c;
#pragma unroll
            for (int i = 0; i < 8; ++i) c[i] = bliv[nt];
#pragma unroll
            for (int kt = 0; kt < 4; ++kt)
                c = wmma_bf16(as[kt], wliF[kt * 4 + nt][l], c);
#pragma unroll
            for (int i = 0; i < 8; ++i) {
                const float v = 0.9f * ml[nt][i] + c[i];
                ml[nt][i] = v;
                acc[nt][i] += sgn * v;
            }
        }

        // ---- window boundary: dp_feat = (end_mean - start_mean)
        if ((t & 31) == 31) {
            const int w = t >> 5;
#pragma unroll
            for (int nt = 0; nt < 4; ++nt)
#pragma unroll
                for (int i = 0; i < 8; ++i) {
                    const int m = i + 8 * hh;
                    dp[(size_t)(b0 + m) * DPF + w * HDIM + nt * 16 + nloc] =
                        acc[nt][i] * (1.0f / 16.0f);
                    acc[nt][i] = 0.0f;
                }
        }
    }
}

// ---------------------------------------------------------------------------
// Kernel 2/3: TDA MLP (tiny)
// ---------------------------------------------------------------------------
__global__ void tda_fc1_kernel(const float* __restrict__ tda,
                               const float* __restrict__ Wt1,
                               const float* __restrict__ bt1,
                               float* __restrict__ h1) {
    const int idx = blockIdx.x * blockDim.x + threadIdx.x;   // B*64
    const int b = idx >> 6, j = idx & 63;
    const float* r = tda + (size_t)b * 150;
    float s = bt1[j];
    for (int k = 0; k < 150; ++k) s += r[k] * Wt1[k * 64 + j];
    h1[(size_t)b * 64 + j] = fmaxf(s, 0.0f);
}

__global__ void tda_fc2_kernel(const float* __restrict__ h1,
                               const float* __restrict__ Wt2,
                               const float* __restrict__ bt2,
                               float* __restrict__ tf) {
    const int idx = blockIdx.x * blockDim.x + threadIdx.x;   // B*32
    const int b = idx >> 5, j = idx & 31;
    const float* r = h1 + (size_t)b * 64;
    float s = bt2[j];
#pragma unroll 8
    for (int k = 0; k < 64; ++k) s += r[k] * Wt2[k * 32 + j];
    tf[(size_t)b * 32 + j] = fmaxf(s, 0.0f);
}

// ---------------------------------------------------------------------------
// Kernel 4: h = concat(dp, tf) @ Wc1 + bc1, plus column sum / sum-sq for BN
// ---------------------------------------------------------------------------
__global__ __launch_bounds__(128) void cls_fc1_kernel(
    const float* __restrict__ dp, const float* __restrict__ tf,
    const float* __restrict__ Wc1, const float* __restrict__ bc1,
    float* __restrict__ hbuf, float* __restrict__ stats) {
    const int b = blockIdx.x;
    const int j = threadIdx.x;                 // 0..127
    const float* dr = dp + (size_t)b * DPF;
    const float* tr = tf + (size_t)b * 32;
    float s = bc1[j];
    for (int k = 0; k < DPF; ++k)   s += dr[k] * Wc1[k * 128 + j];
    for (int k = 0; k < 32; ++k)    s += tr[k] * Wc1[(DPF + k) * 128 + j];
    hbuf[(size_t)b * 128 + j] = s;
    atomicAdd(&stats[j], s);
    atomicAdd(&stats[128 + j], s * s);
}

__global__ void zero_stats_kernel(float* __restrict__ stats) {
    stats[threadIdx.x] = 0.0f;
}

// ---------------------------------------------------------------------------
// Kernel 5: BatchNorm (batch stats) + ReLU + final 128x4 projection
// ---------------------------------------------------------------------------
__global__ void cls_out_kernel(const float* __restrict__ hbuf,
                               const float* __restrict__ stats,
                               const float* __restrict__ gamma,
                               const float* __restrict__ beta,
                               const float* __restrict__ Wc2,
                               const float* __restrict__ bc2,
                               float* __restrict__ out) {
    const int b = blockIdx.x * blockDim.x + threadIdx.x;     // B
    const float invB = 1.0f / (float)BATCH;
    float o0 = bc2[0], o1 = bc2[1], o2 = bc2[2], o3 = bc2[3];
    const float* hr = hbuf + (size_t)b * 128;
    for (int j = 0; j < 128; ++j) {
        const float mu = stats[j] * invB;
        const float var = stats[128 + j] * invB - mu * mu;
        float hv = (hr[j] - mu) * rsqrtf(var + 1e-5f) * gamma[j] + beta[j];
        hv = fmaxf(hv, 0.0f);
        o0 += hv * Wc2[j * 4 + 0];
        o1 += hv * Wc2[j * 4 + 1];
        o2 += hv * Wc2[j * 4 + 2];
        o3 += hv * Wc2[j * 4 + 3];
    }
    out[(size_t)b * 4 + 0] = o0;
    out[(size_t)b * 4 + 1] = o1;
    out[(size_t)b * 4 + 2] = o2;
    out[(size_t)b * 4 + 3] = o3;
}

// ---------------------------------------------------------------------------
extern "C" void kernel_launch(void* const* d_in, const int* in_sizes, int n_in,
                              void* d_out, int out_size, void* d_ws, size_t ws_size,
                              hipStream_t stream) {
    const float* x     = (const float*)d_in[0];
    const float* tda   = (const float*)d_in[1];
    const float* Wsf   = (const float*)d_in[2];
    const float* bsf   = (const float*)d_in[3];
    const float* Wli   = (const float*)d_in[4];
    const float* bli   = (const float*)d_in[5];
    const float* Wt1   = (const float*)d_in[6];
    const float* bt1   = (const float*)d_in[7];
    const float* Wt2   = (const float*)d_in[8];
    const float* bt2   = (const float*)d_in[9];
    const float* Wc1   = (const float*)d_in[10];
    const float* bc1   = (const float*)d_in[11];
    const float* gamma = (const float*)d_in[12];
    const float* beta  = (const float*)d_in[13];
    const float* Wc2   = (const float*)d_in[14];
    const float* bc2   = (const float*)d_in[15];
    float* out = (float*)d_out;

    float* ws   = (float*)d_ws;
    float* dp   = ws;                          // B*640
    float* h1   = dp + (size_t)BATCH * DPF;    // B*64
    float* tf   = h1 + (size_t)BATCH * 64;     // B*32
    float* hbuf = tf + (size_t)BATCH * 32;     // B*128
    float* st   = hbuf + (size_t)BATCH * 128;  // 256

    zero_stats_kernel<<<1, 256, 0, stream>>>(st);
    snn_scan_kernel<<<BATCH / 64, 128, 0, stream>>>(x, Wsf, bsf, Wli, bli, dp);
    tda_fc1_kernel<<<(BATCH * 64) / 256, 256, 0, stream>>>(tda, Wt1, bt1, h1);
    tda_fc2_kernel<<<(BATCH * 32) / 256, 256, 0, stream>>>(h1, Wt2, bt2, tf);
    cls_fc1_kernel<<<BATCH, 128, 0, stream>>>(dp, tf, Wc1, bc1, hbuf, st);
    cls_out_kernel<<<BATCH / 256, 256, 0, stream>>>(hbuf, st, gamma, beta, Wc2, bc2, out);
}
